// SimpleCrossBatchGate_20933670601093
// MI455X (gfx1250) — matrix operand order repaired and checked by reading, not compiled, not run on hardware
//
#include <hip/hip_runtime.h>
#include <hip/hip_bf16.h>

typedef _Float16 v8h  __attribute__((ext_vector_type(8)));
typedef _Float16 v16h __attribute__((ext_vector_type(16)));
typedef float    v8f  __attribute__((ext_vector_type(8)));

#define BATCH 8192
#define HDIM  2048
#define TOPK  32
#define MROWS 32              // rows per block (two 16-row WMMA M-tiles share each B frag)
#define NCHUNK 128            // columns of similarity processed per block iteration
#define NEGF  (-3.402823466e38f)

__device__ __forceinline__ v16h cat8(v8h a, v8h b) {
    return __builtin_shufflevector(a, b, 0,1,2,3,4,5,6,7,8,9,10,11,12,13,14,15);
}

// ---------------------------------------------------------------------------
// Kernel 1: row L2-norm, write normalized rows as f16 (WMMA operand format)
// ---------------------------------------------------------------------------
__global__ __launch_bounds__(256)
void normalize_kernel(const float* __restrict__ h, _Float16* __restrict__ nrm) {
    const int row  = blockIdx.x;
    const int t    = threadIdx.x;
    const int lane = t & 31, wave = t >> 5;
    const float* hrow = h + (size_t)row * HDIM;

    const float4* hp = (const float4*)hrow;
    float4 x0 = hp[t];
    float4 x1 = hp[t + 256];
    float s = x0.x*x0.x + x0.y*x0.y + x0.z*x0.z + x0.w*x0.w
            + x1.x*x1.x + x1.y*x1.y + x1.z*x1.z + x1.w*x1.w;
    #pragma unroll
    for (int off = 16; off; off >>= 1) s += __shfl_xor(s, off, 32);

    __shared__ float red[8];
    __shared__ float sinv;
    if (lane == 0) red[wave] = s;
    __syncthreads();
    if (t == 0) {
        float tot = 0.f;
        #pragma unroll
        for (int i = 0; i < 8; ++i) tot += red[i];
        sinv = 1.0f / fmaxf(sqrtf(tot), 1e-12f);
    }
    __syncthreads();
    const float inv = sinv;

    _Float16* nrow = nrm + (size_t)row * HDIM;
    int c0 = t * 4, c1 = 1024 + t * 4;
    nrow[c0+0] = (_Float16)(x0.x * inv);
    nrow[c0+1] = (_Float16)(x0.y * inv);
    nrow[c0+2] = (_Float16)(x0.z * inv);
    nrow[c0+3] = (_Float16)(x0.w * inv);
    nrow[c1+0] = (_Float16)(x1.x * inv);
    nrow[c1+1] = (_Float16)(x1.y * inv);
    nrow[c1+2] = (_Float16)(x1.z * inv);
    nrow[c1+3] = (_Float16)(x1.w * inv);
}

// ---------------------------------------------------------------------------
// Kernel 2: 32-row stripe per block, all 8192 columns streamed in 128-col
// chunks. Each wave computes TWO 16x16 WMMA M-tiles against ONE register-held
// B fragment (32 flops per L2 byte). A-tile rows are INTERLEAVED in LDS
// (row r at slot (r%16)*2 + r/16) so both M-tiles' fragments are reachable
// from one base register with 16-bit DS immediates (no v_add address
// materialization -> no WAR-hazard v_nops around the WMMAs).
// ---------------------------------------------------------------------------
__global__ __launch_bounds__(256)
void sim_topk_kernel(const _Float16* __restrict__ nrm,
                     const unsigned char* __restrict__ attn,
                     float* __restrict__ outW, int* __restrict__ outI) {
    __shared__ _Float16 Atile[MROWS * HDIM];   // 128 KB (interleaved rows)
    __shared__ float    simC[MROWS * NCHUNK];  // 16 KB
    __shared__ float    tkV[MROWS * TOPK];     // 4 KB
    __shared__ int      tkI[MROWS * TOPK];     // 4 KB

    const int t    = threadIdx.x;
    const int lane = t & 31, wave = t >> 5;
    const int rowBase = blockIdx.x * MROWS;

    // cooperative load of A stripe (32 x 2048 f16) into LDS, interleaved
    {
        const int r = t >> 3;                  // logical row 0..31
        const int c = (t & 7) * 256;           // 0..1792
        const int p = ((r & 15) << 1) | (r >> 4);   // physical slot
        const v8h* src = (const v8h*)(nrm + (size_t)(rowBase + r) * HDIM + c);
        v8h* dst = (v8h*)(Atile + p * HDIM + c);
        #pragma unroll
        for (int i = 0; i < 32; ++i) dst[i] = src[i];
    }
    for (int i = t; i < MROWS * TOPK; i += 256) { tkV[i] = NEGF; tkI[i] = 0; }
    __syncthreads();

    const int m16  = lane & 15;
    const int half = lane >> 4;                // 0 or 1
    const int hiA  = half ? 8 : 0;             // C-matrix M offset for lanes 16-31
    const int aSeg = half * 8;                 // A-frag K sub-segment

    for (int nBase = 0; nBase < BATCH; nBase += NCHUNK) {
        // ---- two 16x16 tiles per wave: cols [nBase+wave*16, +16) ----
        const int j = nBase + wave * 16 + m16;          // B column (= row of N)
        const _Float16* brow = nrm + (size_t)j * HDIM + half * 8;
        const _Float16* arow0 = Atile + (m16 * 2) * HDIM;      // logical row m16
        const _Float16* arow1 = arow0 + HDIM;                  // logical row m16+16

        v8f acc0 = {}, acc1 = {};
        for (int kb = 0; kb < HDIM; kb += 32) {
            v8h b0 = *(const v8h*)(brow + kb);
            v8h b1 = *(const v8h*)(brow + kb + 16);
            v16h bv = cat8(b0, b1);
            v16h av0 = cat8(*(const v8h*)(arow0 + kb + aSeg),
                            *(const v8h*)(arow0 + kb + 16 + aSeg));
            v16h av1 = cat8(*(const v8h*)(arow1 + kb + aSeg),
                            *(const v8h*)(arow1 + kb + 16 + aSeg));
            acc0 = __builtin_amdgcn_wmma_f32_16x16x32_f16(
                false, av0, false, bv, (short)0, acc0, false, false);
            acc1 = __builtin_amdgcn_wmma_f32_16x16x32_f16(
                false, av1, false, bv, (short)0, acc1, false, false);
        }

        // diag + attention mask, spill both tiles to the LDS chunk buffer
        const bool aok = attn[j] != 0;
        #pragma unroll
        for (int v = 0; v < 8; ++v) {
            const int m0 = v + hiA;            // 0..15
            const int m1 = 16 + m0;            // 16..31
            float v0 = acc0[v], v1 = acc1[v];
            if (!aok || (rowBase + m0) == j) v0 = NEGF;
            if (!aok || (rowBase + m1) == j) v1 = NEGF;
            simC[m0 * NCHUNK + wave * 16 + m16] = v0;
            simC[m1 * NCHUNK + wave * 16 + m16] = v1;
        }
        __syncthreads();

        // ---- running top-32 merge: wave w owns rows 4w..4w+3 ----
        #pragma unroll 1
        for (int rr = 0; rr < 4; ++rr) {
            const int row = wave * 4 + rr;
            // chunk max for early-out
            float cmax = NEGF;
            #pragma unroll
            for (int i = 0; i < 4; ++i)
                cmax = fmaxf(cmax, simC[row * NCHUNK + lane * 4 + i]);
            #pragma unroll
            for (int off = 16; off; off >>= 1)
                cmax = fmaxf(cmax, __shfl_xor(cmax, off, 32));
            if (cmax <= tkV[row * TOPK + (TOPK - 1)]) continue;

            // snapshot 160 candidates (32 old topk + 128 chunk) into registers
            float cv[5]; int ci[5];
            #pragma unroll
            for (int s = 0; s < 5; ++s) {
                const int c = lane * 5 + s;                // 0..159
                if (c < TOPK) { cv[s] = tkV[row * TOPK + c]; ci[s] = tkI[row * TOPK + c]; }
                else { const int col = c - TOPK;
                       cv[s] = simC[row * NCHUNK + col];   ci[s] = nBase + col; }
            }
            // 32x max-extract (descending order; index-ascending tie break)
            for (int it = 0; it < TOPK; ++it) {
                float v = cv[0]; int sl = 0;
                #pragma unroll
                for (int s = 1; s < 5; ++s) if (cv[s] > v) { v = cv[s]; sl = s; }
                int p = ci[sl], ln = lane;
                #pragma unroll
                for (int off = 16; off; off >>= 1) {
                    float ov = __shfl_xor(v, off, 32);
                    int   op = __shfl_xor(p, off, 32);
                    int   ol = __shfl_xor(ln, off, 32);
                    int   os = __shfl_xor(sl, off, 32);
                    if (ov > v || (ov == v && op < p)) { v = ov; p = op; ln = ol; sl = os; }
                }
                if (lane == 0) { tkV[row * TOPK + it] = v; tkI[row * TOPK + it] = p; }
                #pragma unroll
                for (int s = 0; s < 5; ++s)
                    if (lane == ln && sl == s) cv[s] = NEGF;
            }
        }
        __syncthreads();
    }

    // ---- softmax over the 32 survivors, write weights + indices ----
    #pragma unroll 1
    for (int rr = 0; rr < 4; ++rr) {
        const int row  = wave * 4 + rr;
        const int rowG = rowBase + row;
        const float v  = tkV[row * TOPK + lane];
        const float mx = tkV[row * TOPK];           // sorted: slot 0 is max
        float e = __expf(v - mx);
        float s = e;
        #pragma unroll
        for (int off = 16; off; off >>= 1) s += __shfl_xor(s, off, 32);
        outW[(size_t)rowG * TOPK + lane] = e / s;
        outI[(size_t)rowG * TOPK + lane] = tkI[row * TOPK + lane];
    }
}

// ---------------------------------------------------------------------------
// Kernel 3: sparse cross = sum_j w_j * H[idx_j], gate = sigmoid(<[h,cross],gw>+b),
//           out = h + gate*cross    (one pass, gathers hit L2: H is 64 MB)
// ---------------------------------------------------------------------------
__global__ __launch_bounds__(256)
void cross_gate_kernel(const float* __restrict__ h,
                       const float* __restrict__ wgt, const int* __restrict__ idx,
                       const float* __restrict__ gw,  const float* __restrict__ gb,
                       float* __restrict__ out) {
    const int row  = blockIdx.x;
    const int t    = threadIdx.x;
    const int lane = t & 31, wave = t >> 5;
    const int c0   = t * 8;

    __shared__ float sw[TOPK];
    __shared__ int   sid[TOPK];
    __shared__ float red[8];
    __shared__ float sg;
    if (t < TOPK) { sw[t] = wgt[(size_t)row * TOPK + t]; sid[t] = idx[(size_t)row * TOPK + t]; }
    __syncthreads();

    float cr[8] = {0.f,0.f,0.f,0.f,0.f,0.f,0.f,0.f};
    #pragma unroll 4
    for (int j = 0; j < TOPK; ++j) {
        const float wj = sw[j];
        const float4* src = (const float4*)(h + (size_t)sid[j] * HDIM + c0);
        float4 a = src[0], b = src[1];
        cr[0] += wj * a.x; cr[1] += wj * a.y; cr[2] += wj * a.z; cr[3] += wj * a.w;
        cr[4] += wj * b.x; cr[5] += wj * b.y; cr[6] += wj * b.z; cr[7] += wj * b.w;
    }

    // gate partial: gw[0:2048]*h + gw[2048:4096]*cross
    const float4* hp  = (const float4*)(h  + (size_t)row * HDIM + c0);
    const float4* g1p = (const float4*)(gw + c0);
    const float4* g2p = (const float4*)(gw + HDIM + c0);
    float4 h0 = hp[0],  h1 = hp[1];
    float4 ga = g1p[0], gc = g1p[1];
    float4 gb0 = g2p[0], gb1 = g2p[1];
    float part = h0.x*ga.x + h0.y*ga.y + h0.z*ga.z + h0.w*ga.w
               + h1.x*gc.x + h1.y*gc.y + h1.z*gc.z + h1.w*gc.w
               + cr[0]*gb0.x + cr[1]*gb0.y + cr[2]*gb0.z + cr[3]*gb0.w
               + cr[4]*gb1.x + cr[5]*gb1.y + cr[6]*gb1.z + cr[7]*gb1.w;
    #pragma unroll
    for (int off = 16; off; off >>= 1) part += __shfl_xor(part, off, 32);
    if (lane == 0) red[wave] = part;
    __syncthreads();
    if (t == 0) {
        float tot = 0.f;
        #pragma unroll
        for (int i = 0; i < 8; ++i) tot += red[i];
        sg = 1.0f / (1.0f + __expf(-(tot + gb[0])));
    }
    __syncthreads();
    const float g = sg;

    float* orow = out + (size_t)row * HDIM + c0;
    orow[0] = h0.x + g * cr[0]; orow[1] = h0.y + g * cr[1];
    orow[2] = h0.z + g * cr[2]; orow[3] = h0.w + g * cr[3];
    orow[4] = h1.x + g * cr[4]; orow[5] = h1.y + g * cr[5];
    orow[6] = h1.z + g * cr[6]; orow[7] = h1.w + g * cr[7];
}

// ---------------------------------------------------------------------------
extern "C" void kernel_launch(void* const* d_in, const int* in_sizes, int n_in,
                              void* d_out, int out_size, void* d_ws, size_t ws_size,
                              hipStream_t stream) {
    const float*         hidden = (const float*)d_in[0];
    const unsigned char* attn   = (const unsigned char*)d_in[1];
    const float*         gate_w = (const float*)d_in[2];
    const float*         gate_b = (const float*)d_in[3];
    float*               out    = (float*)d_out;

    // workspace: [ normalized f16 (32MB) | weights f32 (1MB) | indices i32 (1MB) ]
    char* ws = (char*)d_ws;
    _Float16* nrm  = (_Float16*)ws;
    float*    wgt  = (float*)(ws + (size_t)BATCH * HDIM * sizeof(_Float16));
    int*      tidx = (int*)(ws + (size_t)BATCH * HDIM * sizeof(_Float16)
                               + (size_t)BATCH * TOPK * sizeof(float));

    normalize_kernel<<<BATCH, 256, 0, stream>>>(hidden, nrm);
    sim_topk_kernel<<<BATCH / MROWS, 256, 0, stream>>>(nrm, attn, wgt, tidx);
    cross_gate_kernel<<<BATCH, 256, 0, stream>>>(hidden, wgt, tidx, gate_w, gate_b, out);
}
